// GCN_81776177316393
// MI455X (gfx1250) — matrix operand order repaired and checked by reading, not compiled, not run on hardware
//
#include <hip/hip_runtime.h>
#include <hip/hip_bf16.h>

// ---------------------------------------------------------------------------
// MI455X (gfx1250): bf16 WMMA GEMMs (v_wmma_f32_16x16x32_bf16, f32 accum) with
// GLOBAL_LOAD_ASYNC_TO_LDS_B128 double-buffered tile staging (ASYNCcnt).
// wave32, 8 waves / 256-thread block, 128x128x32 tiles.
// ---------------------------------------------------------------------------

typedef __attribute__((ext_vector_type(16))) __bf16 bf16x16;
typedef __attribute__((ext_vector_type(8)))  __bf16 bf16x8;
typedef __attribute__((ext_vector_type(8)))  float  f32x8;
typedef __attribute__((ext_vector_type(4)))  int    i32x4;

typedef __attribute__((address_space(1))) char  gchar;
typedef __attribute__((address_space(3))) char  lchar;
typedef __attribute__((address_space(1))) i32x4 g_i32x4;   // global int4
typedef __attribute__((address_space(3))) i32x4 l_i32x4;   // LDS int4

static __device__ __forceinline__ __bf16 f2bf(float f) {
    unsigned u = __builtin_bit_cast(unsigned, f);
    u += 0x7FFFu + ((u >> 16) & 1u);            // round-to-nearest-even
    unsigned short s = (unsigned short)(u >> 16);
    return __builtin_bit_cast(__bf16, s);
}

#define BM 128
#define BN 128
#define BK 32
#define LDSROW 80                 // 32 bf16 (64B) + 16B pad, keeps 16B align
#define TILEB  (128 * LDSROW)     // 10240 B per tile

enum { OM_F32 = 0, OM_BF16 = 1, OM_BF16T = 2 };

// Generic tiled GEMM:  C[M][N] = act[M][K](bf16, row-major)  x  Bt[N][K](bf16)
// CONV=true adds the 3-tap conv accumulation (row shift +/-1 in t, T=512,
// zero-padded) with per-tap weight planes Bt + tap*N*K.
// OM selects epilogue: f32 row-major, bf16 row-major, or bf16 transposed
// grouped by 256 rows ([row/256][n][row%256]) to feed the next GEMM's B side.
template<bool CONV, int OM, bool RELU>
__global__ __launch_bounds__(256)
void gemm_wmma_bf16(const __bf16* __restrict__ A, const __bf16* __restrict__ B,
                    const float* __restrict__ bias, void* __restrict__ outp,
                    int M, int N, int K, long sAb, long sBb, long sOb)
{
    __shared__ __align__(16) char smem[4 * TILEB];   // A0,B0 | A1,B1

    const int bz = blockIdx.z;
    // byte-granular address-space-typed bases for the async copy engine
    gchar* Ag = (gchar*)(void*)(A + (long)bz * sAb);
    gchar* Bg = (gchar*)(void*)(B + (long)bz * sBb);
    lchar* Ls = (lchar*)(void*)smem;

    const int tid  = threadIdx.x;
    const int lane = tid & 31;
    const int wave = tid >> 5;
    const int wm   = (wave & 1) * 64;     // 2 waves across M (64 rows each)
    const int wn   = (wave >> 1) * 32;    // 4 waves across N (32 cols each)
    const int m0   = blockIdx.y * BM;
    const int n0   = blockIdx.x * BN;
    const int lrow = lane & 15;
    const int lh   = lane >> 4;

    const int ksteps = K >> 5;
    const int S = (CONV ? 3 : 1) * ksteps;

    f32x8 acc[4][2];
#pragma unroll
    for (int fm = 0; fm < 4; ++fm)
#pragma unroll
        for (int fn = 0; fn < 2; ++fn)
#pragma unroll
            for (int v = 0; v < 8; ++v) acc[fm][fn][v] = 0.0f;

    // Issue async global->LDS copies for pipeline step s (no data VGPRs).
    auto issue = [&](int s) {
        const int tap = CONV ? (s / ksteps) : 0;
        const int ks  = CONV ? (s - tap * ksteps) : s;
        const int kb  = ks << 5;                       // K element base
        const int p   = s & 1;
        gchar* Bt = Bg + (CONV ? (long)tap * (long)N * (long)K * 2 : 0l);
        lchar* sA = Ls + p * (2 * TILEB);
        lchar* sB = sA + TILEB;
#pragma unroll
        for (int i = 0; i < 2; ++i) {
            const int idx = tid + (i << 8);            // 512 16B-chunks/tile
            const int row = idx >> 2;
            const int cb  = idx & 3;
            const int gm  = m0 + row;
            const long lofs = (long)row * LDSROW + (cb << 4);
            bool valid = true;
            long srow  = gm;
            if (CONV) {                                // tap shift, zero pad
                const int ts = (gm & 511) + tap - 1;
                valid = (ts >= 0) && (ts < 512);
                srow  = (long)gm + (tap - 1);
            }
            if (valid) {
                __builtin_amdgcn_global_load_async_to_lds_b128(
                    (g_i32x4*)(Ag + (srow * (long)K + kb + (cb << 3)) * 2),
                    (l_i32x4*)(sA + lofs), 0, 0);
            } else {
                uint4 z; z.x = 0u; z.y = 0u; z.z = 0u; z.w = 0u;
                *(uint4*)(smem + p * (2 * TILEB) + lofs) = z;  // ds_store zeros
            }
            __builtin_amdgcn_global_load_async_to_lds_b128(
                (g_i32x4*)(Bt + ((long)(n0 + row) * (long)K + kb + (cb << 3)) * 2),
                (l_i32x4*)(sB + lofs), 0, 0);
        }
    };

    issue(0);
    for (int s = 0; s < S; ++s) {
        const int p = s & 1;
        const char* sA = smem + p * (2 * TILEB);
        const char* sB = sA + TILEB;

        asm volatile("s_wait_asynccnt 0x0" ::: "memory");  // step-s tiles landed
        __syncthreads();                                    // ... in all waves

        if (s + 1 < S) issue(s + 1);   // async copies overlap the WMMA block

        bf16x16 af[4];
#pragma unroll
        for (int fm = 0; fm < 4; ++fm) {
            // A frag: lane row = lrow; K halves 0..7 / 8..15 by lane half,
            // then +16 for VGPRs 4..7 (ISA 16-bit A layout).
            const char* base = sA + (wm + fm * 16 + lrow) * LDSROW + lh * 16;
            bf16x8 h0 = *(const bf16x8*)(base);
            bf16x8 h1 = *(const bf16x8*)(base + 32);
            af[fm] = __builtin_shufflevector(h0, h1,
                     0,1,2,3,4,5,6,7,8,9,10,11,12,13,14,15);
        }
#pragma unroll
        for (int fn = 0; fn < 2; ++fn) {
            // B frag: lane column = lrow; K 8*lh..+7, then 16+8*lh..+7 (matches A).
            const char* base = sB + (wn + fn * 16 + lrow) * LDSROW + lh * 16;
            bf16x8 h0 = *(const bf16x8*)(base);
            bf16x8 h1 = *(const bf16x8*)(base + 32);
            bf16x16 bfv = __builtin_shufflevector(h0, h1,
                          0,1,2,3,4,5,6,7,8,9,10,11,12,13,14,15);
#pragma unroll
            for (int fm = 0; fm < 4; ++fm)
                acc[fm][fn] = __builtin_amdgcn_wmma_f32_16x16x32_bf16(
                    false, af[fm], false, bfv, (short)0, acc[fm][fn],
                    false, false);
        }
        __syncthreads();   // all waves done with buf p before it is re-issued
    }

    // epilogue: C layout lane=(n%16), vgpr v -> m = v + 8*(lane/16)
    float*  of = (float*)outp  + (long)bz * sOb;
    __bf16* oh = (__bf16*)outp + (long)bz * sOb;
#pragma unroll
    for (int fn = 0; fn < 2; ++fn) {
        const int gn = n0 + wn + fn * 16 + lrow;
        const float bv = bias ? bias[gn] : 0.0f;
#pragma unroll
        for (int fm = 0; fm < 4; ++fm) {
            const long gmb = m0 + wm + fm * 16 + (lh << 3);
#pragma unroll
            for (int v = 0; v < 8; ++v) {
                const long gm = gmb + v;
                float val = acc[fm][fn][v] + bv;
                if (RELU) val = fmaxf(val, 0.0f);
                if (OM == OM_F32)
                    of[gm * (long)N + gn] = val;
                else if (OM == OM_BF16)
                    oh[gm * (long)N + gn] = f2bf(val);
                else // OM_BF16T: [gm/256][n][gm%256]
                    oh[(((gm >> 8) * (long)N + gn) << 8) + (gm & 255)] = f2bf(val);
            }
        }
    }
}

// ----------------------- preprocessing kernels ------------------------------

__global__ void k_cvt_bf16(const float* __restrict__ in, __bf16* __restrict__ out, long n)
{
    long i = (long)blockIdx.x * blockDim.x + threadIdx.x;
    const long stride = (long)gridDim.x * blockDim.x;
    for (; i < n; i += stride) out[i] = f2bf(in[i]);
}

// conv_w OIH [2048][2048][3] -> wk[k][o][i] bf16
__global__ void k_cvt_convw(const float* __restrict__ w, __bf16* __restrict__ wk)
{
    const long n = 3l * 2048 * 2048;
    long i = (long)blockIdx.x * blockDim.x + threadIdx.x;
    const long stride = (long)gridDim.x * blockDim.x;
    for (; i < n; i += stride) {
        const long k   = i / (2048l * 2048);
        const long rem = i - k * 2048l * 2048;
        const long o   = rem >> 11;
        const long c   = rem & 2047;
        wk[i] = f2bf(w[((o << 11) + c) * 3 + k]);
    }
}

// W [in=2048][out=2048] -> Wt[out][in] bf16
__global__ void k_cvt_transpose(const float* __restrict__ w, __bf16* __restrict__ wt)
{
    const long n = 2048l * 2048;
    long i = (long)blockIdx.x * blockDim.x + threadIdx.x;
    const long stride = (long)gridDim.x * blockDim.x;
    for (; i < n; i += stride) {
        const long o = i >> 11;
        const long c = i & 2047;
        wt[i] = f2bf(w[(c << 11) + o]);
    }
}

// a = max(adj, adj^T) + I ; row-normalize D^-1 A ; store bf16
__global__ __launch_bounds__(256)
void k_norm_adj(const float* __restrict__ adj, __bf16* __restrict__ anorm)
{
    __shared__ float red[256];
    const int b = blockIdx.x >> 8;
    const int i = blockIdx.x & 255;
    const int j = threadIdx.x;
    const float* ab = adj + (long)b * 65536;
    const float aij = ab[i * 256 + j];
    const float aji = ab[j * 256 + i];
    float a = (aji > aij) ? aji : aij;     // == adj + at*mask - adj*mask
    if (i == j) a += 1.0f;
    red[j] = a;
    __syncthreads();
    for (int s2 = 128; s2 > 0; s2 >>= 1) {
        if (j < s2) red[j] += red[j + s2];
        __syncthreads();
    }
    const float sum = red[0];
    const float inv = sum > 0.0f ? 1.0f / sum : 0.0f;
    anorm[(long)b * 65536 + i * 256 + j] = f2bf(a * inv);
}

// ------------------------------ launcher ------------------------------------

extern "C" void kernel_launch(void* const* d_in, const int* in_sizes, int n_in,
                              void* d_out, int out_size, void* d_ws, size_t ws_size,
                              hipStream_t stream)
{
    const float* x      = (const float*)d_in[0];   // [4,3,512,2048]
    const float* nodes  = (const float*)d_in[1];   // [4,256,2048]
    const float* adj    = (const float*)d_in[2];   // [4,256,256]
    const float* conv_w = (const float*)d_in[3];   // [2048,2048,3] OIH
    const float* conv_b = (const float*)d_in[4];   // [2048]
    const float* W1     = (const float*)d_in[5];   // [2048,2048]
    const float* b1     = (const float*)d_in[6];
    const float* W2     = (const float*)d_in[7];
    const float* b2     = (const float*)d_in[8];
    float* out = (float*)d_out;                    // feats (12.58M) ++ out (2.10M)

    char* ws = (char*)d_ws;
    auto carve = [&](size_t bytes) {
        char* p = ws;
        ws += (bytes + 255) & ~(size_t)255;
        return p;
    };
    __bf16* xb     = (__bf16*)carve(6144l * 2048 * 2);      // bf16 x, [6144][2048]
    __bf16* wk     = (__bf16*)carve(3l * 2048 * 2048 * 2);  // [k][o][i]
    __bf16* w1t    = (__bf16*)carve(2048l * 2048 * 2);      // W1^T [out][in]
    __bf16* w2t    = (__bf16*)carve(2048l * 2048 * 2);      // W2^T
    __bf16* nodesb = (__bf16*)carve(1024l * 2048 * 2);      // [4*256][2048]
    __bf16* anorm  = (__bf16*)carve(4l * 256 * 256 * 2);    // [4][256][256]
    __bf16* z1t    = (__bf16*)carve(4l * 2048 * 256 * 2);   // [4][feat][node]
    __bf16* hbuf   = (__bf16*)carve(1024l * 2048 * 2);      // [4*256][2048]
    __bf16* z2t    = (__bf16*)carve(4l * 2048 * 256 * 2);   // [4][feat][node]

    // -- precision / layout conversions ------------------------------------
    k_cvt_bf16     <<<4096, 256, 0, stream>>>(x, xb, 6144l * 2048);
    k_cvt_convw    <<<4096, 256, 0, stream>>>(conv_w, wk);
    k_cvt_transpose<<<4096, 256, 0, stream>>>(W1, w1t);
    k_cvt_transpose<<<4096, 256, 0, stream>>>(W2, w2t);
    k_cvt_bf16     <<<2048, 256, 0, stream>>>(nodes, nodesb, 1024l * 2048);
    k_norm_adj     <<<1024, 256, 0, stream>>>(adj, anorm);

    // -- Conv1d as 3-tap GEMM: feats = relu(x * Wk + b), M=6144,N=2048,K=2048
    gemm_wmma_bf16<true, OM_F32, true><<<dim3(16, 48, 1), 256, 0, stream>>>(
        xb, wk, conv_b, out, 6144, 2048, 2048, 0, 0, 0);

    // -- z1^T = (nodes @ W1)^T, stored [b][feat][node] bf16
    gemm_wmma_bf16<false, OM_BF16T, false><<<dim3(16, 8, 1), 256, 0, stream>>>(
        nodesb, w1t, nullptr, z1t, 1024, 2048, 2048, 0, 0, 0);

    // -- h = relu(anorm @ z1 + b1), batched over 4, M=256,N=2048,K=256
    gemm_wmma_bf16<false, OM_BF16, true><<<dim3(16, 2, 4), 256, 0, stream>>>(
        anorm, z1t, b1, hbuf, 256, 2048, 256,
        65536l, 2048l * 256, 256l * 2048);

    // -- z2^T = (h @ W2)^T
    gemm_wmma_bf16<false, OM_BF16T, false><<<dim3(16, 8, 1), 256, 0, stream>>>(
        hbuf, w2t, nullptr, z2t, 1024, 2048, 2048, 0, 0, 0);

    // -- out = anorm @ z2 + b2 (f32, no relu)
    gemm_wmma_bf16<false, OM_F32, false><<<dim3(16, 2, 4), 256, 0, stream>>>(
        anorm, z2t, b2, out + 6144l * 2048, 256, 2048, 256,
        65536l, 2048l * 256, 256l * 2048);
}
